// MultiLayerEnhancedPGATActor_28329604284639
// MI455X (gfx1250) — matrix operand order repaired and checked by reading, not compiled, not run on hardware
//
#include <hip/hip_runtime.h>

// ---------------------------------------------------------------------------
// MultiLayerEnhancedPGATActor — fully folded, WMMA-based, memory-bound kernel
// for gfx1250 (MI455X).  All linear algebra bracketing the two softmaxes is
// folded at launch time; the streaming kernel does
//   qk0 = qf(19) @ Mcat0(32x32)          [WMMA f16, K=32, N=32]
//   m0  = softmax-mix (36 vals + bias 1)
//   qk1 = m0ext(37) @ F1(64x32)          [WMMA f16, K=64, N=32]
//   m1  = softmax-mix
//   out = m1ext(37) @ GH(37x4)           [VALU]
// Roofline: ~21 MB streamed @ 23.3 TB/s ≈ 0.9 us floor; ~0.6 GFLOP total.
// ---------------------------------------------------------------------------

typedef _Float16 h16 __attribute__((ext_vector_type(16)));
typedef _Float16 h8  __attribute__((ext_vector_type(8)));
typedef float    f8  __attribute__((ext_vector_type(8)));
typedef float    f4  __attribute__((ext_vector_type(4)));

// Intra-wave LDS producer->consumer handoff: wave32 lanes are lockstep, we only
// need the DS counter drained + a compiler scheduling fence (no workgroup barrier).
#define WSYNC() do { asm volatile("s_wait_dscnt 0" ::: "memory"); \
                     __builtin_amdgcn_wave_barrier(); } while (0)

__device__ __forceinline__ f8 wmma_f16(h16 a, h16 b, f8 c) {
  return __builtin_amdgcn_wmma_f32_16x16x32_f16(false, a, false, b, (short)0, c,
                                                false, false);
}

// A fragment (16x32 f16) from row-major LDS rows[16][64]:
// lanes 0-15: M=lane,   elems 0..7 -> K = kt*32+0..7,  elems 8..15 -> K = kt*32+16..23
// lanes16-31: M=lane-16, same +8.  Both halves are contiguous 16B runs.
__device__ __forceinline__ h16 ldA(const _Float16* rowbase, int lane, int kt) {
  const _Float16* p = rowbase + kt * 32 + ((lane >> 4) << 3);
  union { h16 v; h8 q[2]; } u;
  u.q[0] = *(const h8*)(p);
  u.q[1] = *(const h8*)(p + 16);
  return u.v;
}

// B fragment: pre-swizzled so each lane reads 16 contiguous halves (2x b128).
__device__ __forceinline__ h16 ldB(const _Float16* mat, int tileIdx, int lane) {
  const _Float16* p = mat + tileIdx * 512 + lane * 16;
  union { h16 v; h8 q[2]; } u;
  u.q[0] = *(const h8*)(p);
  u.q[1] = *(const h8*)(p + 8);
  return u.v;
}

// Per-lane softmax/mixing for 2 heads (h0, h0+1).  qrow = 32 qk values in LDS.
// sw* already include the 1/sqrt(32) attention scale.
// 2-way softmax as sigmoid: one v_exp_f32 + one v_rcp_f32 per softmax.
__device__ __forceinline__ void attend_m(
    const float* qrow,
    const float nb0[6], const float nb1[6], float e70, float e71,
    float obx0, float oby0, float obx1, float oby1,
    float swa0, float swa1, float swo0, float swo1,
    int h0, float ma[2][7], float mo[2][2]) {
#pragma unroll
  for (int t = 0; t < 2; ++t) {
    const int h = h0 + t;
    float d0 = 0.f, d1 = 0.f;
#pragma unroll
    for (int j = 0; j < 6; ++j) {
      const float qv = qrow[j * 4 + h];
      d0 += nb0[j] * qv;
      d1 += nb1[j] * qv;
    }
    const float ez = __expf(d1 * swa1 - d0 * swa0);   // exp(l1 - l0)
    const float p0 = __builtin_amdgcn_rcpf(1.0f + ez);
    const float p1 = ez * p0;
#pragma unroll
    for (int j = 0; j < 6; ++j) ma[t][j] = p0 * nb0[j] + p1 * nb1[j];
    ma[t][6] = p0 * e70 + p1 * e71;

    const float g0 = (obx0 * qrow[24 + h] + oby0 * qrow[28 + h]) * swo0;
    const float g1 = (obx1 * qrow[24 + h] + oby1 * qrow[28 + h]) * swo1;
    const float fz = __expf(g1 - g0);
    const float q0 = __builtin_amdgcn_rcpf(1.0f + fz);
    const float q1 = fz * q0;
    mo[t][0] = q0 * obx0 + q1 * obx1;
    mo[t][1] = q0 * oby0 + q1 * oby1;
  }
}

// ---------------------------------------------------------------------------
// Main streaming kernel: 256 threads = 8 waves; each wave owns 16 samples.
// ---------------------------------------------------------------------------
__global__ __launch_bounds__(256) void actor_fused(
    const float* __restrict__ obs,
    const _Float16* __restrict__ gBQ0,   // swizzled Mcat0  (32x32  -> 2 B-tiles)
    const _Float16* __restrict__ gBQ1,   // swizzled F1     (64x32  -> 4 B-tiles)
    const float* __restrict__ gGH,       // GH f32 [37][4]
    const float* __restrict__ pca0, const float* __restrict__ pco0,
    const float* __restrict__ pca1, const float* __restrict__ pco1,
    float* __restrict__ outp) {
  __shared__ __align__(32) _Float16 sBQ0[1024];
  __shared__ __align__(32) _Float16 sBQ1[2048];
  __shared__ float sGH[148];
  __shared__ __align__(32) _Float16 sRowA[8][16][64];  // A staging (f16)
  __shared__ float sQK[8][16][40];                     // qk scatter / m1 xchg
  __shared__ __align__(16) float sObs[8][592];         // 16 obs rows per wave

  const float kScale = 0.17677669529663687f;  // (C/H)^-0.5 = 1/sqrt(32)

  const int tid = threadIdx.x;
  // cooperative: folded weights -> LDS (once per workgroup, shared by 8 waves)
  for (int i = tid; i < 512; i += 256)
    ((unsigned int*)sBQ0)[i] = ((const unsigned int*)gBQ0)[i];
  for (int i = tid; i < 1024; i += 256)
    ((unsigned int*)sBQ1)[i] = ((const unsigned int*)gBQ1)[i];
  if (tid < 148) sGH[tid] = gGH[tid];
  const float vca0 = *pca0, vco0 = *pco0, vca1 = *pca1, vco1 = *pco1;
  __syncthreads();

  const int w    = tid >> 5;
  const int lane = tid & 31;
  const int sloc = lane & 15;          // sample within tile (2 lanes/sample)
  const int h0   = (lane >> 4) << 1;   // this lane handles heads h0, h0+1
  const int tile = blockIdx.x * 8 + w; // 8192 tiles of 16 samples

  // ---- stage 16 obs rows (contiguous 2368B) coalesced via b128 ----
  {
    const f4* src = (const f4*)(obs + (size_t)tile * 592);
    f4* dst = (f4*)(&sObs[w][0]);
#pragma unroll
    for (int i = 0; i < 5; ++i) {
      int k = lane + i * 32;
      if (k < 148) dst[k] = src[k];
    }
  }
  WSYNC();

  // ---- per-sample features + qf staging (A for qk0) ----
  const float* orow = &sObs[w][sloc * 37];
  float nb0[6], nb1[6];
#pragma unroll
  for (int j = 0; j < 6; ++j) { nb0[j] = orow[12 + j]; nb1[j] = orow[18 + j]; }
  const float ax = orow[0], ay = orow[1];
  const float e70 = orow[24], e71 = orow[25];
  const float obx0 = orow[26], oby0 = orow[27];
  const float obx1 = orow[28], oby1 = orow[29];

  _Float16* rowW = &sRowA[w][sloc][0];
  const _Float16* rowR = rowW;
  if (lane < 16) {
#pragma unroll
    for (int i = 0; i < 12; ++i) rowW[i] = (_Float16)orow[i];
#pragma unroll
    for (int i = 12; i < 16; ++i) rowW[i] = (_Float16)orow[30 + i - 12];
  } else {
#pragma unroll
    for (int i = 16; i < 19; ++i) rowW[i] = (_Float16)orow[30 + i - 12];
#pragma unroll
    for (int i = 19; i < 32; ++i) rowW[i] = (_Float16)0.0f;
  }
  // distances (layer-independent); raw v_sqrt_f32 is plenty (feeds exp())
  float dx, dy;
  dx = ax - nb0[0]; dy = ay - nb0[1];
  const float dA0 = __builtin_amdgcn_sqrtf(dx * dx + dy * dy);
  dx = ax - nb1[0]; dy = ay - nb1[1];
  const float dA1 = __builtin_amdgcn_sqrtf(dx * dx + dy * dy);
  dx = ax - obx0;   dy = ay - oby0;
  const float dO0 = __builtin_amdgcn_sqrtf(dx * dx + dy * dy);
  dx = ax - obx1;   dy = ay - oby1;
  const float dO1 = __builtin_amdgcn_sqrtf(dx * dx + dy * dy);
  WSYNC();

  // ---- qk0 = qf @ Mcat0 : WMMA 16x16x32, N-tiles 0,1 ----
  {
    h16 a0 = ldA(rowR, lane, 0);
    f8 c0 = {}; f8 c1 = {};
    c0 = wmma_f16(a0, ldB(sBQ0, 0, lane), c0);
    c1 = wmma_f16(a0, ldB(sBQ0, 1, lane), c1);
    const int ro = (lane >> 4) << 3;  // D layout: M = r + 8*(lane/16), N = lane%16
#pragma unroll
    for (int r2 = 0; r2 < 8; ++r2) {
      sQK[w][ro + r2][sloc] = c0[r2];
      sQK[w][ro + r2][16 + sloc] = c1[r2];
    }
  }
  WSYNC();

  // ---- layer 0 softmax/mix -> m0 (A for qk1, bias row appended) ----
  {
    const float swa0 = kScale * __expf(-vca0 * dA0);
    const float swa1 = kScale * __expf(-vca0 * dA1);
    const float swo0 = kScale * __expf(-vco0 * dO0);
    const float swo1 = kScale * __expf(-vco0 * dO1);
    float ma[2][7], mo[2][2];
    attend_m(&sQK[w][sloc][0], nb0, nb1, e70, e71, obx0, oby0, obx1, oby1,
             swa0, swa1, swo0, swo1, h0, ma, mo);
#pragma unroll
    for (int t = 0; t < 2; ++t) {
      const int h = h0 + t;
#pragma unroll
      for (int j = 0; j < 7; ++j) rowW[j * 4 + h] = (_Float16)ma[t][j];
      rowW[28 + h] = (_Float16)mo[t][0];
      rowW[32 + h] = (_Float16)mo[t][1];
    }
    if (lane < 16) {
      rowW[36] = (_Float16)1.0f;  // bias row of F1
#pragma unroll
      for (int i = 37; i < 50; ++i) rowW[i] = (_Float16)0.0f;
    } else {
#pragma unroll
      for (int i = 50; i < 64; ++i) rowW[i] = (_Float16)0.0f;
    }
  }
  WSYNC();

  // ---- qk1 = m0ext @ F1 : WMMA, K = 64 (2 steps), N-tiles 0,1 ----
  {
    h16 a0 = ldA(rowR, lane, 0);
    h16 a1 = ldA(rowR, lane, 1);
    f8 c0 = {}; f8 c1 = {};
    c0 = wmma_f16(a0, ldB(sBQ1, 0, lane), c0);
    c0 = wmma_f16(a1, ldB(sBQ1, 2, lane), c0);
    c1 = wmma_f16(a0, ldB(sBQ1, 1, lane), c1);
    c1 = wmma_f16(a1, ldB(sBQ1, 3, lane), c1);
    const int ro = (lane >> 4) << 3;
#pragma unroll
    for (int r2 = 0; r2 < 8; ++r2) {
      sQK[w][ro + r2][sloc] = c0[r2];
      sQK[w][ro + r2][16 + sloc] = c1[r2];
    }
  }
  WSYNC();

  // ---- layer 1 softmax/mix -> m1 (exchanged through LDS, f32) ----
  {
    const float swa0 = kScale * __expf(-vca1 * dA0);
    const float swa1 = kScale * __expf(-vca1 * dA1);
    const float swo0 = kScale * __expf(-vco1 * dO0);
    const float swo1 = kScale * __expf(-vco1 * dO1);
    float ma[2][7], mo[2][2];
    attend_m(&sQK[w][sloc][0], nb0, nb1, e70, e71, obx0, oby0, obx1, oby1,
             swa0, swa1, swo0, swo1, h0, ma, mo);
    float* r = &sQK[w][sloc][0];
#pragma unroll
    for (int t = 0; t < 2; ++t) {
      const int h = h0 + t;
#pragma unroll
      for (int j = 0; j < 7; ++j) r[j * 4 + h] = ma[t][j];
      r[28 + h] = mo[t][0];
      r[32 + h] = mo[t][1];
    }
  }
  WSYNC();

  // ---- head: out = m1ext @ GH (36x4 + bias row) ----
  if (lane < 16) {
    const float* mm = &sQK[w][lane][0];
    float a0 = sGH[144], a1 = sGH[145], a2 = sGH[146], a3 = sGH[147];
#pragma unroll
    for (int i = 0; i < 36; ++i) {
      const float v = mm[i];
      a0 += v * sGH[i * 4 + 0];
      a1 += v * sGH[i * 4 + 1];
      a2 += v * sGH[i * 4 + 2];
      a3 += v * sGH[i * 4 + 3];
    }
    f4 res = {a0, a1, a2, a3};
    *(f4*)(outp + (size_t)(tile * 16 + lane) * 4) = res;
  }
}

// ---------------------------------------------------------------------------
// Launch-time weight folding (tiny, runs on stream before the main kernel).
// ---------------------------------------------------------------------------

// Mcat[d,(j,h)] = sum_c wq[d,h*128+c] * wk[j,h*128+c]; cols 0..23 = wak, 24..31 = wok
__global__ void pk_mcat(const float* __restrict__ wq, const float* __restrict__ wak,
                        const float* __restrict__ wok, int dq, int Kp,
                        float* __restrict__ out) {
  int idx = blockIdx.x * blockDim.x + threadIdx.x;
  if (idx >= Kp * 32) return;
  int K = idx >> 5, col = idx & 31;
  float v = 0.f;
  if (K < dq) {
    int j, h; const float* wk;
    if (col < 24) { j = col >> 2; h = col & 3; wk = wak; }
    else          { j = (col - 24) >> 2; h = (col - 24) & 3; wk = wok; }
    const float* a = wq + K * 512 + h * 128;
    const float* b = wk + j * 512 + h * 128;
    for (int c = 0; c < 128; ++c) v += a[c] * b[c];
  }
  out[idx] = v;
}

// PAcat[(j,h),c'] : rows 0..27 = wav x apw (per head), rows 28..35 = wov x opw
__global__ void pk_pacat(const float* __restrict__ wav, const float* __restrict__ wov,
                         const float* __restrict__ apw, const float* __restrict__ opw,
                         float* __restrict__ out) {
  int idx = blockIdx.x * blockDim.x + threadIdx.x;
  if (idx >= 36 * 128) return;
  int row = idx >> 7, c2 = idx & 127;
  float v = 0.f;
  if (row < 28) {
    int j = row >> 2, h = row & 3;
    const float* wv = wav + j * 512 + h * 128;
    const float* pw = apw + (h * 128) * 128 + c2;
    for (int c = 0; c < 128; ++c) v += wv[c] * pw[c * 128];
  } else {
    int j = (row - 28) >> 2, h = (row - 28) & 3;
    const float* wv = wov + j * 512 + h * 128;
    const float* pw = opw + (h * 128) * 128 + c2;
    for (int c = 0; c < 128; ++c) v += wv[c] * pw[c * 128];
  }
  out[idx] = v;
}

// Gext[37][128]: rows 0..27 via fpw top half, 28..35 via bottom half, 36 = folded bias
__global__ void pk_gext(const float* __restrict__ pa, const float* __restrict__ fpw,
                        const float* __restrict__ apb, const float* __restrict__ opb,
                        const float* __restrict__ fpb, float* __restrict__ out) {
  int idx = blockIdx.x * blockDim.x + threadIdx.x;
  if (idx >= 37 * 128) return;
  int row = idx / 128, c2 = idx % 128;
  float v = 0.f;
  if (row < 28) {
    for (int cp = 0; cp < 128; ++cp) v += pa[row * 128 + cp] * fpw[cp * 128 + c2];
  } else if (row < 36) {
    for (int cp = 0; cp < 128; ++cp) v += pa[row * 128 + cp] * fpw[(128 + cp) * 128 + c2];
  } else {
    for (int cp = 0; cp < 128; ++cp)
      v += apb[cp] * fpw[cp * 128 + c2] + opb[cp] * fpw[(128 + cp) * 128 + c2];
    v += fpb[c2];
  }
  out[idx] = v;
}

// F1[64][32] = G0ext(37x128) @ Mcat1(128x32), zero-padded to K=64
__global__ void pk_f1(const float* __restrict__ g0, const float* __restrict__ mc1,
                      float* __restrict__ out) {
  int idx = blockIdx.x * blockDim.x + threadIdx.x;
  if (idx >= 64 * 32) return;
  int row = idx >> 5, col = idx & 31;
  float v = 0.f;
  if (row < 37)
    for (int d = 0; d < 128; ++d) v += g0[row * 128 + d] * mc1[d * 32 + col];
  out[idx] = v;
}

// GH[37][4] = G1ext @ headw (+ headb folded into bias row 36)
__global__ void pk_gh(const float* __restrict__ g1, const float* __restrict__ headw,
                      const float* __restrict__ headb, float* __restrict__ out) {
  int idx = threadIdx.x;
  if (idx >= 148) return;
  int row = idx >> 2, o = idx & 3;
  float v = (row == 36) ? headb[o] : 0.f;
  for (int c2 = 0; c2 < 128; ++c2) v += g1[row * 128 + c2] * headw[c2 * 4 + o];
  out[idx] = v;
}

// f32 [Kp][32] -> f16 B fragments in the exact 32x16 wave32 lane layout:
// within tile (kt,nt): lane L = 16*((kin/8)%2)+nin, elem e = kin%8 + 8*(kin/16)
__global__ void pk_swizzleB(const float* __restrict__ src, int Kp,
                            _Float16* __restrict__ dst) {
  int idx = blockIdx.x * blockDim.x + threadIdx.x;
  if (idx >= Kp * 32) return;
  int K = idx >> 5, n = idx & 31;
  int kt = K >> 5, kin = K & 31;
  int nt = n >> 4, nin = n & 15;
  int L = (((kin >> 3) & 1) << 4) + nin;
  int e = (kin & 7) + ((kin >> 4) << 3);
  dst[((kt * 2 + nt) * 32 + L) * 16 + e] = (_Float16)src[idx];
}

// ---------------------------------------------------------------------------
extern "C" void kernel_launch(void* const* d_in, const int* in_sizes, int n_in,
                              void* d_out, int out_size, void* d_ws, size_t ws_size,
                              hipStream_t stream) {
  (void)in_sizes; (void)n_in; (void)out_size; (void)ws_size;
  const float* obs   = (const float*)d_in[0];
  const float* wq0   = (const float*)d_in[1];
  const float* wak0  = (const float*)d_in[2];
  const float* wav0  = (const float*)d_in[3];
  const float* wok0  = (const float*)d_in[4];
  const float* wov0  = (const float*)d_in[5];
  const float* apw0  = (const float*)d_in[6];
  const float* apb0  = (const float*)d_in[7];
  const float* opw0  = (const float*)d_in[8];
  const float* opb0  = (const float*)d_in[9];
  const float* fpw0  = (const float*)d_in[10];
  const float* fpb0  = (const float*)d_in[11];
  const float* ca0   = (const float*)d_in[12];
  const float* co0   = (const float*)d_in[13];
  const float* wq1   = (const float*)d_in[14];
  const float* wak1  = (const float*)d_in[15];
  const float* wav1  = (const float*)d_in[16];
  const float* wok1  = (const float*)d_in[17];
  const float* wov1  = (const float*)d_in[18];
  const float* apw1  = (const float*)d_in[19];
  const float* apb1  = (const float*)d_in[20];
  const float* opw1  = (const float*)d_in[21];
  const float* opb1  = (const float*)d_in[22];
  const float* fpw1  = (const float*)d_in[23];
  const float* fpb1  = (const float*)d_in[24];
  const float* ca1   = (const float*)d_in[25];
  const float* co1   = (const float*)d_in[26];
  const float* headw = (const float*)d_in[27];
  const float* headb = (const float*)d_in[28];

  // ws layout (floats): ~110 KB total
  float* wsf    = (float*)d_ws;
  float* mcat0f = wsf + 0;      // [32][32]
  float* mcat1f = wsf + 1024;   // [128][32]
  float* pa0    = wsf + 5120;   // [36][128]
  float* g0     = wsf + 9728;   // [37][128]
  float* f1f    = wsf + 14464;  // [64][32]
  float* pa1    = wsf + 16512;  // [36][128]
  float* g1     = wsf + 21120;  // [37][128]
  float* gh     = wsf + 25856;  // [37][4]
  _Float16* bq0 = (_Float16*)(wsf + 26112);  // 1024 halves
  _Float16* bq1 = (_Float16*)(wsf + 26624);  // 2048 halves

  pk_mcat <<<4, 256, 0, stream>>>(wq0, wak0, wok0, 19, 32, mcat0f);
  pk_mcat <<<16, 256, 0, stream>>>(wq1, wak1, wok1, 128, 128, mcat1f);
  pk_pacat<<<18, 256, 0, stream>>>(wav0, wov0, apw0, opw0, pa0);
  pk_gext <<<19, 256, 0, stream>>>(pa0, fpw0, apb0, opb0, fpb0, g0);
  pk_f1   <<<8, 256, 0, stream>>>(g0, mcat1f, f1f);
  pk_pacat<<<18, 256, 0, stream>>>(wav1, wov1, apw1, opw1, pa1);
  pk_gext <<<19, 256, 0, stream>>>(pa1, fpw1, apb1, opb1, fpb1, g1);
  pk_gh   <<<1, 160, 0, stream>>>(g1, headw, headb, gh);
  pk_swizzleB<<<4, 256, 0, stream>>>(mcat0f, 32, bq0);
  pk_swizzleB<<<8, 256, 0, stream>>>(f1f, 64, bq1);

  // 131072 samples = 1024 blocks x 8 waves x 16 samples
  actor_fused<<<1024, 256, 0, stream>>>(obs, bq0, bq1, gh,
                                        ca0, co0, ca1, co1, (float*)d_out);
}
